// InformerTrendReturn_35347580846626
// MI455X (gfx1250) — matrix operand order repaired and checked by reading, not compiled, not run on hardware
//
#include <hip/hip_runtime.h>
#include <hip/hip_bf16.h>
#include <math.h>

#define B_    8
#define S_    4096
#define F_    22
#define D_    128
#define H_    256
#define NB_   5
#define L_    3
#define KTOP_ 81
#define EPS_  1e-5f

typedef __bf16 bf16_t;
typedef __attribute__((ext_vector_type(16))) __bf16 v16bf;
typedef __attribute__((ext_vector_type(8)))  __bf16 v8bf;
typedef __attribute__((ext_vector_type(8)))  float  v8f;

// D = A(16x32 bf16) * B(32x16 bf16) + C(16x16 f32)
__device__ __forceinline__ v8f wmma_bf16(v16bf a, v16bf b, v8f c) {
  return __builtin_amdgcn_wmma_f32_16x16x32_bf16(false, a, false, b, (short)0, c,
                                                 false, false);
}

// A fragment: element (m, k). src is row-major [m][k], row stride ld (elems).
// Lane l holds m=l&15, K = kbase..kbase+7 and kbase+16..kbase+23, kbase=(l>>4)*8.
__device__ __forceinline__ v16bf load_A(const bf16_t* src, int ld, int kb) {
  int l  = threadIdx.x & 31;
  int m  = l & 15;
  int ko = (l >> 4) * 8;
  const bf16_t* p = src + m * ld + kb + ko;
  v8bf lo = *(const v8bf*)(p);
  v8bf hi = *(const v8bf*)(p + 16);
  v16bf r;
#pragma unroll
  for (int i = 0; i < 8; ++i) { r[i] = lo[i]; r[i + 8] = hi[i]; }
  return r;
}

// B fragment: element (k, n). src indexed [n][k] (k contiguous), row stride ld.
// Lane l holds n=l&15, K = kbase..kbase+7 and kbase+16..kbase+23, kbase=(l>>4)*8.
__device__ __forceinline__ v16bf load_B(const bf16_t* src, int ld, int kb) {
  int l  = threadIdx.x & 31;
  int n  = l & 15;
  int k0 = kb + (l >> 4) * 8;
  const bf16_t* p = src + n * ld + k0;
  v8bf lo = *(const v8bf*)(p);
  v8bf hi = *(const v8bf*)(p + 16);
  v16bf r;
#pragma unroll
  for (int i = 0; i < 8; ++i) { r[i] = lo[i]; r[i + 8] = hi[i]; }
  return r;
}

// -------------------- embed + pos + input LN --------------------
__global__ void __launch_bounds__(256)
k_embed(const float* __restrict__ x, const float* __restrict__ eW,
        const float* __restrict__ eb, const float* __restrict__ gam,
        const float* __restrict__ bet, const float* __restrict__ pos,
        float* __restrict__ hf, bf16_t* __restrict__ hb, bf16_t* __restrict__ hbT) {
  int w = threadIdx.x >> 5, lane = threadIdx.x & 31;
  int row = blockIdx.x * 8 + w;     // 0..B*S-1
  int b = row / S_, s = row % S_;
  float v[4];
#pragma unroll
  for (int j = 0; j < 4; ++j) {
    int d = lane + 32 * j;
    float acc = eb[d] + pos[s * D_ + d];
    for (int f = 0; f < F_; ++f) acc += x[(size_t)row * F_ + f] * eW[f * D_ + d];
    v[j] = acc;
  }
  float sum = v[0] + v[1] + v[2] + v[3];
#pragma unroll
  for (int off = 16; off > 0; off >>= 1) sum += __shfl_xor(sum, off, 32);
  float mean = sum * (1.0f / D_);
  float var = 0.f;
#pragma unroll
  for (int j = 0; j < 4; ++j) { float dd = v[j] - mean; var += dd * dd; }
#pragma unroll
  for (int off = 16; off > 0; off >>= 1) var += __shfl_xor(var, off, 32);
  float rstd = rsqrtf(var * (1.0f / D_) + EPS_);
#pragma unroll
  for (int j = 0; j < 4; ++j) {
    int d = lane + 32 * j;
    float y = (v[j] - mean) * rstd * gam[d] + bet[d];
    hf[(size_t)row * D_ + d]  = y;
    hb[(size_t)row * D_ + d]  = (bf16_t)y;
    hbT[((size_t)b * D_ + d) * S_ + s] = (bf16_t)y;
  }
}

// -------------------- prob-sparse self-attention --------------------
// One block = 16 query rows of one batch, 8 waves.
// Pass1: row max (keys split across waves).
// Pass2: 256-bin histogram of (max - s) -> top-81 threshold.
// Pass3 (cooperative): 256-key chunks; each wave computes exp-weights for its
// 32-key slice into shared LDS, then all waves WMMA the shared weight tile
// against V^T for their private 16-dim output slice. Z reduced across waves.
__global__ void __launch_bounds__(256)
k_attn(const bf16_t* __restrict__ hb, const bf16_t* __restrict__ hbT,
       float* __restrict__ att) {
  const float rsD = 0.08838834764831845f;  // 1/sqrt(128)
  const float RANGE = 16.0f, INV_BINW = 16.0f, BINW = 0.0625f;
  int w = threadIdx.x >> 5, lane = threadIdx.x & 31;
  int half = lane >> 4, nn = lane & 15;
  int b  = blockIdx.x / (S_ / 16);
  int q0 = (blockIdx.x % (S_ / 16)) * 16;
  const bf16_t* Q  = hb  + ((size_t)b * S_ + q0) * D_;
  const bf16_t* K  = hb  + (size_t)b * S_ * D_;
  const bf16_t* VT = hbT + (size_t)b * D_ * S_;

  __shared__ float    s_pmax[8][16];
  __shared__ float    s_max[16];
  __shared__ float    s_thr[16];
  __shared__ float    s_pz[8][16];
  __shared__ float    s_z[16];
  __shared__ unsigned s_hist[16][256];
  __shared__ __align__(32) bf16_t s_wc[16][256];   // shared weight tile, 8KB

  for (int i = threadIdx.x; i < 16 * 256; i += 256) ((unsigned*)s_hist)[i] = 0u;

  v16bf qa[4];
#pragma unroll
  for (int c = 0; c < 4; ++c) qa[c] = load_A(Q, D_, c * 32);

  // ---- pass 1: row max ----
  float vmax[8];
#pragma unroll
  for (int g0 = 0; g0 < 8; ++g0) vmax[g0] = -3.0e38f;
  for (int kt = 0; kt < 32; ++kt) {
    int kb = w * 512 + kt * 16;
    if (kt + 1 < 32)
      __builtin_prefetch(K + (size_t)(kb + 16) * D_, 0, 3);  // global_prefetch_b8
    v8f acc;
#pragma unroll
    for (int g0 = 0; g0 < 8; ++g0) acc[g0] = 0.f;
#pragma unroll
    for (int c = 0; c < 4; ++c)
      acc = wmma_bf16(qa[c], load_B(K + (size_t)kb * D_, D_, c * 32), acc);
#pragma unroll
    for (int g0 = 0; g0 < 8; ++g0) vmax[g0] = fmaxf(vmax[g0], acc[g0] * rsD);
  }
#pragma unroll
  for (int g0 = 0; g0 < 8; ++g0)
#pragma unroll
    for (int off = 1; off < 16; off <<= 1)
      vmax[g0] = fmaxf(vmax[g0], __shfl_xor(vmax[g0], off, 32));
  if (nn == 0)
#pragma unroll
    for (int g0 = 0; g0 < 8; ++g0) s_pmax[w][g0 + 8 * half] = vmax[g0];
  __syncthreads();
  if (threadIdx.x < 16) {
    float m = -3.0e38f;
    for (int ww = 0; ww < 8; ++ww) m = fmaxf(m, s_pmax[ww][threadIdx.x]);
    s_max[threadIdx.x] = m;
  }
  __syncthreads();

  // ---- pass 2: histogram of (max - s) ----
  for (int kt = 0; kt < 32; ++kt) {
    int kb = w * 512 + kt * 16;
    v8f acc;
#pragma unroll
    for (int g0 = 0; g0 < 8; ++g0) acc[g0] = 0.f;
#pragma unroll
    for (int c = 0; c < 4; ++c)
      acc = wmma_bf16(qa[c], load_B(K + (size_t)kb * D_, D_, c * 32), acc);
#pragma unroll
    for (int g0 = 0; g0 < 8; ++g0) {
      int row = g0 + 8 * half;
      float s = acc[g0] * rsD;
      int bin = (int)((s_max[row] - s) * INV_BINW);
      bin = bin < 0 ? 0 : (bin > 255 ? 255 : bin);
      atomicAdd(&s_hist[row][bin], 1u);
    }
  }
  __syncthreads();
  if (threadIdx.x < 16) {
    unsigned cum = 0;
    float m = s_max[threadIdx.x];
    float thr = m - RANGE;
    for (int bin = 0; bin < 256; ++bin) {
      cum += s_hist[threadIdx.x][bin];
      if (cum >= KTOP_) { thr = m - (float)(bin + 1) * BINW; break; }
    }
    s_thr[threadIdx.x] = thr;
  }
  __syncthreads();

  // ---- pass 3 (cooperative): weights + AV + Z ----
  int dc = w * 16;                       // this wave's output dim slice
  v8f oacc;
  float zacc[8];
#pragma unroll
  for (int g0 = 0; g0 < 8; ++g0) { oacc[g0] = 0.f; zacc[g0] = 0.f; }
  for (int kc = 0; kc < 16; ++kc) {      // 256-key chunks
    // each wave: scores + weights for its 32-key slice of the chunk
#pragma unroll
    for (int sub = 0; sub < 2; ++sub) {
      int ks = kc * 256 + w * 32 + sub * 16;
      v8f acc;
#pragma unroll
      for (int g0 = 0; g0 < 8; ++g0) acc[g0] = 0.f;
#pragma unroll
      for (int c = 0; c < 4; ++c)
        acc = wmma_bf16(qa[c], load_B(K + (size_t)ks * D_, D_, c * 32), acc);
#pragma unroll
      for (int g0 = 0; g0 < 8; ++g0) {
        int row = g0 + 8 * half;
        float s = acc[g0] * rsD;
        float e = (s >= s_thr[row]) ? __expf(s - s_max[row]) : 0.0f;
        zacc[g0] += e;
        s_wc[row][w * 32 + sub * 16 + nn] = (bf16_t)e;
      }
    }
    __syncthreads();
    // all waves: AV over this 256-key chunk for their dim slice
#pragma unroll
    for (int t = 0; t < 8; ++t) {
      v16bf aw = load_A(&s_wc[0][0], 256, t * 32);
      v16bf bv = load_B(VT + (size_t)dc * S_, S_, kc * 256 + t * 32);
      oacc = wmma_bf16(aw, bv, oacc);
    }
    __syncthreads();
  }
  // Z: reduce across lanes, then across waves
#pragma unroll
  for (int g0 = 0; g0 < 8; ++g0)
#pragma unroll
    for (int off = 1; off < 16; off <<= 1) zacc[g0] += __shfl_xor(zacc[g0], off, 32);
  if (nn == 0)
#pragma unroll
    for (int g0 = 0; g0 < 8; ++g0) s_pz[w][g0 + 8 * half] = zacc[g0];
  __syncthreads();
  if (threadIdx.x < 16) {
    float z = 0.f;
    for (int ww = 0; ww < 8; ++ww) z += s_pz[ww][threadIdx.x];
    s_z[threadIdx.x] = z;
  }
  __syncthreads();
#pragma unroll
  for (int g0 = 0; g0 < 8; ++g0) {
    int m = g0 + 8 * half;
    att[((size_t)b * S_ + q0 + m) * D_ + dc + nn] = oacc[g0] / s_z[m];
  }
}

// -------------------- h = LN(h + res) --------------------
__global__ void __launch_bounds__(256)
k_resln(const float* __restrict__ res, float* __restrict__ hf,
        const float* __restrict__ gam, const float* __restrict__ bet,
        bf16_t* __restrict__ hb, bf16_t* __restrict__ hbT) {
  int w = threadIdx.x >> 5, lane = threadIdx.x & 31;
  int row = blockIdx.x * 8 + w;
  int b = row / S_, s = row % S_;
  float v[4];
#pragma unroll
  for (int j = 0; j < 4; ++j) {
    int d = lane + 32 * j;
    v[j] = hf[(size_t)row * D_ + d] + res[(size_t)row * D_ + d];
  }
  float sum = v[0] + v[1] + v[2] + v[3];
#pragma unroll
  for (int off = 16; off > 0; off >>= 1) sum += __shfl_xor(sum, off, 32);
  float mean = sum * (1.0f / D_);
  float var = 0.f;
#pragma unroll
  for (int j = 0; j < 4; ++j) { float dd = v[j] - mean; var += dd * dd; }
#pragma unroll
  for (int off = 16; off > 0; off >>= 1) var += __shfl_xor(var, off, 32);
  float rstd = rsqrtf(var * (1.0f / D_) + EPS_);
#pragma unroll
  for (int j = 0; j < 4; ++j) {
    int d = lane + 32 * j;
    float y = (v[j] - mean) * rstd * gam[d] + bet[d];
    hf[(size_t)row * D_ + d] = y;
    hb[(size_t)row * D_ + d] = (bf16_t)y;
    hbT[((size_t)b * D_ + d) * S_ + s] = (bf16_t)y;
  }
}

// -------------------- FFN + residual LN (16 rows per block) --------------------
__global__ void __launch_bounds__(256)
k_ffn(float* __restrict__ hf, bf16_t* __restrict__ hb, bf16_t* __restrict__ hbT,
      const bf16_t* __restrict__ W1t, const float* __restrict__ b1,
      const bf16_t* __restrict__ W2t, const float* __restrict__ b2,
      const float* __restrict__ g2, const float* __restrict__ be2) {
  int w = threadIdx.x >> 5, lane = threadIdx.x & 31;
  int half = lane >> 4, nn = lane & 15;
  int r0 = blockIdx.x * 16;
  const bf16_t* X = hb + (size_t)r0 * D_;

  __shared__ __align__(32) bf16_t s_t1[16 * H_];   // relu(X@W1+b1), bf16
  __shared__ float  s_t2[16 * D_];                 // T1@W2+b2

  // stage 1: wave handles T1 cols [w*32, w*32+32)
#pragma unroll
  for (int cc = 0; cc < 2; ++cc) {
    int nb = w * 32 + cc * 16;
    float bias = b1[nb + nn];
    v8f acc;
#pragma unroll
    for (int g0 = 0; g0 < 8; ++g0) acc[g0] = bias;
#pragma unroll
    for (int c = 0; c < 4; ++c)
      acc = wmma_bf16(load_A(X, D_, c * 32), load_B(W1t + (size_t)nb * D_, D_, c * 32), acc);
#pragma unroll
    for (int g0 = 0; g0 < 8; ++g0) {
      int row = g0 + 8 * half;
      float y = acc[g0] > 0.f ? acc[g0] : 0.f;
      s_t1[row * H_ + nb + nn] = (bf16_t)y;
    }
  }
  __syncthreads();
  // stage 2: wave handles T2 cols [w*16, w*16+16)
  {
    int nb = w * 16;
    float bias = b2[nb + nn];
    v8f acc;
#pragma unroll
    for (int g0 = 0; g0 < 8; ++g0) acc[g0] = bias;
#pragma unroll
    for (int c = 0; c < 8; ++c)
      acc = wmma_bf16(load_A(s_t1, H_, c * 32), load_B(W2t + (size_t)nb * H_, H_, c * 32), acc);
#pragma unroll
    for (int g0 = 0; g0 < 8; ++g0) {
      int row = g0 + 8 * half;
      s_t2[row * D_ + nb + nn] = acc[g0];
    }
  }
  __syncthreads();
  // stage 3: LN(h + T2); wave handles rows 2w, 2w+1
#pragma unroll
  for (int rr = 0; rr < 2; ++rr) {
    int row = w * 2 + rr;
    int grow = r0 + row;
    int b = grow / S_, s = grow % S_;
    float v[4];
#pragma unroll
    for (int j = 0; j < 4; ++j) {
      int d = lane + 32 * j;
      v[j] = hf[(size_t)grow * D_ + d] + s_t2[row * D_ + d];
    }
    float sum = v[0] + v[1] + v[2] + v[3];
#pragma unroll
    for (int off = 16; off > 0; off >>= 1) sum += __shfl_xor(sum, off, 32);
    float mean = sum * (1.0f / D_);
    float var = 0.f;
#pragma unroll
    for (int j = 0; j < 4; ++j) { float dd = v[j] - mean; var += dd * dd; }
#pragma unroll
    for (int off = 16; off > 0; off >>= 1) var += __shfl_xor(var, off, 32);
    float rstd = rsqrtf(var * (1.0f / D_) + EPS_);
#pragma unroll
    for (int j = 0; j < 4; ++j) {
      int d = lane + 32 * j;
      float y = (v[j] - mean) * rstd * g2[d] + be2[d];
      hf[(size_t)grow * D_ + d] = y;
      hb[(size_t)grow * D_ + d] = (bf16_t)y;
      hbT[((size_t)b * D_ + d) * S_ + s] = (bf16_t)y;
    }
  }
}

// -------------------- weight transpose -> bf16 --------------------
__global__ void k_prep(const float* __restrict__ W1, const float* __restrict__ W2,
                       bf16_t* __restrict__ W1t, bf16_t* __restrict__ W2t) {
  int t = blockIdx.x * 256 + threadIdx.x;
  int n = L_ * D_ * H_;
  if (t < n) {
    int l = t / (D_ * H_);
    int r = t % (D_ * H_);
    int o = r / D_, i = r % D_;      // W1t[l][o<H][i<D]
    W1t[t] = (bf16_t)W1[((size_t)l * D_ + i) * H_ + o];
    int o2 = r / H_, i2 = r % H_;    // W2t[l][o2<D][i2<H]
    W2t[t] = (bf16_t)W2[((size_t)l * H_ + i2) * D_ + o2];
  }
}

// -------------------- heads --------------------
__global__ void k_heads(const float* __restrict__ hf, const float* __restrict__ tW,
                        const float* __restrict__ tb, const float* __restrict__ bW,
                        const float* __restrict__ bb, float* __restrict__ out) {
  int t = threadIdx.x;
  if (t < 8) {
    const float* last = hf + ((size_t)t * S_ + (S_ - 1)) * D_;
    float acc = tb[0];
    for (int d = 0; d < D_; ++d) acc += last[d] * tW[d];
    out[t] = 1.0f / (1.0f + __expf(-acc));
  } else if (t < 48) {
    int q = t - 8;
    int b = q / NB_, j = q % NB_;
    const float* last = hf + ((size_t)b * S_ + (S_ - 1)) * D_;
    float acc = bb[j];
    for (int d = 0; d < D_; ++d) acc += last[d] * bW[d * NB_ + j];
    out[8 + q] = acc;
  }
}

extern "C" void kernel_launch(void* const* d_in, const int* in_sizes, int n_in,
                              void* d_out, int out_size, void* d_ws, size_t ws_size,
                              hipStream_t stream) {
  const float* x     = (const float*)d_in[0];
  const float* eW    = (const float*)d_in[1];
  const float* eb    = (const float*)d_in[2];
  const float* ng    = (const float*)d_in[3];
  const float* nbta  = (const float*)d_in[4];
  const float* pos   = (const float*)d_in[5];
  const float* ln1g  = (const float*)d_in[6];
  const float* ln1b  = (const float*)d_in[7];
  const float* W1    = (const float*)d_in[8];
  const float* b1    = (const float*)d_in[9];
  const float* W2    = (const float*)d_in[10];
  const float* b2    = (const float*)d_in[11];
  const float* ln2g  = (const float*)d_in[12];
  const float* ln2b  = (const float*)d_in[13];
  const float* tW    = (const float*)d_in[14];
  const float* tb    = (const float*)d_in[15];
  const float* bW    = (const float*)d_in[16];
  const float* bb    = (const float*)d_in[17];

  char* ws = (char*)d_ws;
  const size_t HF_BYTES  = (size_t)B_ * S_ * D_ * sizeof(float);   // 16 MB
  const size_t HB_BYTES  = (size_t)B_ * S_ * D_ * sizeof(bf16_t);  // 8 MB
  float*  hf  = (float*)(ws);
  float*  att = (float*)(ws + HF_BYTES);
  bf16_t* hb  = (bf16_t*)(ws + 2 * HF_BYTES);
  bf16_t* hbT = (bf16_t*)(ws + 2 * HF_BYTES + HB_BYTES);
  bf16_t* W1t = (bf16_t*)(ws + 2 * HF_BYTES + 2 * HB_BYTES);
  bf16_t* W2t = W1t + (size_t)L_ * D_ * H_;

  k_prep<<<(L_ * D_ * H_ + 255) / 256, 256, 0, stream>>>(W1, W2, W1t, W2t);
  k_embed<<<B_ * S_ / 8, 256, 0, stream>>>(x, eW, eb, ng, nbta, pos, hf, hb, hbT);
  for (int l = 0; l < L_; ++l) {
    k_attn<<<B_ * S_ / 16, 256, 0, stream>>>(hb, hbT, att);
    k_resln<<<B_ * S_ / 8, 256, 0, stream>>>(att, hf, ln1g + l * D_, ln1b + l * D_, hb, hbT);
    k_ffn<<<B_ * S_ / 16, 256, 0, stream>>>(hf, hb, hbT,
                                            W1t + (size_t)l * D_ * H_, b1 + l * H_,
                                            W2t + (size_t)l * D_ * H_, b2 + l * D_,
                                            ln2g + l * D_, ln2b + l * D_);
  }
  k_heads<<<1, 256, 0, stream>>>(hf, tW, tb, bW, bb, (float*)d_out);
}
